// EdgeAwareGATLayer_4621384810754
// MI455X (gfx1250) — compile-verified, matrix operand order checked by reading, not run on hardware
//
#include <hip/hip_runtime.h>
#include <hip/hip_bf16.h>
#include <math.h>

// ---------------- problem constants (from reference) ----------------
#define N_NODES 50000
#define N_EDGES 800000
#define F_IN    256
#define F_OUT   32
#define H_HEADS 8
#define F_EDGE  5
#define C_OUT   256        // H_HEADS * F_OUT
#define A_LEN   69         // 2*F_OUT + F_EDGE
#define LEAKY   0.2f

#define AS1 __attribute__((address_space(1)))
#define AS3 __attribute__((address_space(3)))

typedef __attribute__((ext_vector_type(16))) _Float16 v16h;
typedef __attribute__((ext_vector_type(8)))  _Float16 v8h;
typedef __attribute__((ext_vector_type(8)))  float    v8f;
typedef int b128i __attribute__((vector_size(16)));   // matches builtin's V4i

// LDS B-tile row stride in halves: 264 halves = 528 B.
// Bank step per row = 528/4 = 132 dwords -> 132 % 64 = 4 banks; 16 consecutive
// rows (one fragment's lanes) cover 16*4 = 64 banks -> conflict-free.
#define LDS_BSTRIDE 264

// ---------------- workspace layout (byte offsets) ----------------
static const size_t OFF_WH   = 0;                                           // N*256 f32
static const size_t OFF_WT   = OFF_WH   + (size_t)N_NODES * C_OUT * 4;      // 256*256 f16
static const size_t OFF_SSRC = OFF_WT   + (size_t)256 * 256 * 2;
static const size_t OFF_SDST = OFF_SSRC + (size_t)N_NODES * H_HEADS * 4;
static const size_t OFF_EBUF = OFF_SDST + (size_t)N_NODES * H_HEADS * 4;    // E*8 f32
static const size_t OFF_NMAX = OFF_EBUF + (size_t)N_EDGES * H_HEADS * 4;
static const size_t OFF_NSUM = OFF_NMAX + (size_t)N_NODES * H_HEADS * 4;
static const size_t OFF_CMAT = OFF_NSUM + (size_t)N_NODES * H_HEADS * 4;

// ---------------- prep: pack W -> WT (f16, transposed), fold edge mats ----------------
__global__ __launch_bounds__(256) void prep_kernel(const float* __restrict__ W,
                                                   const float* __restrict__ a,
                                                   const float* __restrict__ Wew,
                                                   _Float16* __restrict__ WT,
                                                   float* __restrict__ Cmat) {
  int idx = blockIdx.x * 256 + threadIdx.x;
  if (idx < 256 * 256) {
    int c = idx >> 8;          // output column = h*32+o
    int f = idx & 255;         // input feature
    int h = c >> 5, o = c & 31;
    WT[(size_t)c * F_IN + f] = (_Float16)W[(size_t)h * F_IN * F_OUT + (size_t)f * F_OUT + o];
  }
  if (blockIdx.x == 0 && threadIdx.x < F_EDGE * H_HEADS) {
    int k = threadIdx.x >> 3;  // 0..4
    int h = threadIdx.x & 7;   // 0..7
    float s = 0.f;
    for (int j = 0; j < F_EDGE; ++j)
      s += Wew[j * F_EDGE + k] * a[h * A_LEN + 2 * F_OUT + j];
    Cmat[k * H_HEADS + h] = s;
  }
}

// ---------------- GEMM: Wh = x @ WT^T via v_wmma_f32_16x16x32_f16 ----------------
// Block: 256 thr = 8 waves. Wave w: row-tile = w>>1 (16 rows), col half = w&1 (128 cols = 8 tiles).
// WT (256x256 f16) staged once per block into LDS via async global->LDS (ASYNCcnt).
__global__ __launch_bounds__(256) void gemm_wh_kernel(const float* __restrict__ x,
                                                      const _Float16* __restrict__ WT,
                                                      float* __restrict__ Wh) {
  __shared__ _Float16 ldsB[256 * LDS_BSTRIDE];   // 135,168 B of the WGP's 320 KB

  // ---- stage B into LDS: 8192 chunks of 16 B (256 rows x 32 chunks)
  {
    const int tid = threadIdx.x;
    #pragma unroll
    for (int i = 0; i < 32; ++i) {
      int c   = i * 256 + tid;
      int row = c >> 5, ch = c & 31;
      const _Float16* g = WT + (size_t)row * F_IN + ch * 8;
      _Float16*       l = &ldsB[row * LDS_BSTRIDE + ch * 8];
#if __has_builtin(__builtin_amdgcn_global_load_async_to_lds_b128)
      __builtin_amdgcn_global_load_async_to_lds_b128(
          (AS1 b128i*)(uintptr_t)g,               // global VA, AS1
          (AS3 b128i*)(unsigned)(uintptr_t)l,     // LDS offset = addr[31:0] (ISA flat rule)
          0, 0);
#else
      *(float4*)l = *(const float4*)g;
#endif
    }
#if __has_builtin(__builtin_amdgcn_global_load_async_to_lds_b128)
#if __has_builtin(__builtin_amdgcn_s_wait_asynccnt)
    __builtin_amdgcn_s_wait_asynccnt(0);
#else
    asm volatile("s_wait_asynccnt 0x0" ::: "memory");
#endif
#endif
    __syncthreads();
  }

  const int lane = threadIdx.x & 31;
  const int wave = threadIdx.x >> 5;
  const int row0 = blockIdx.x * 64 + (wave >> 1) * 16;
  const int colbase = (wave & 1) * 128;
  const int m  = lane & 15;
  const int hi = lane >> 4;          // 0: K{0..7,16..23}  1: K{8..15,24..31}
  const int ka = hi ? 8 : 0;

  int arow = row0 + m;
  if (arow >= N_NODES) arow = N_NODES - 1;   // clamp; stores are guarded
  const float* xrow = x + (size_t)arow * F_IN;

  v8f acc[8] = {};

  #pragma unroll
  for (int kk = 0; kk < 8; ++kk) {
    const int kb = kk * 32;
    if (kk < 7) __builtin_prefetch(xrow + kb + 32, 0, 3);   // WGP-scope prefetch of next A
    // ---- A fragment: 16 f32 -> 16 f16 per lane (ISA 16-bit A 16x32 layout)
    v16h afrag;
    {
      const float4* p0 = (const float4*)(xrow + kb + ka);
      const float4* p1 = (const float4*)(xrow + kb + 16 + ka);
      float4 u0 = p0[0], u1 = p0[1], u2 = p1[0], u3 = p1[1];
      afrag[0]=(_Float16)u0.x; afrag[1]=(_Float16)u0.y; afrag[2]=(_Float16)u0.z; afrag[3]=(_Float16)u0.w;
      afrag[4]=(_Float16)u1.x; afrag[5]=(_Float16)u1.y; afrag[6]=(_Float16)u1.z; afrag[7]=(_Float16)u1.w;
      afrag[8]=(_Float16)u2.x; afrag[9]=(_Float16)u2.y; afrag[10]=(_Float16)u2.z; afrag[11]=(_Float16)u2.w;
      afrag[12]=(_Float16)u3.x; afrag[13]=(_Float16)u3.y; afrag[14]=(_Float16)u3.z; afrag[15]=(_Float16)u3.w;
    }
    // ---- load all 8 B fragments from LDS (conflict-free), then burst 8 WMMAs
    v16h bf[8];
    #pragma unroll
    for (int t = 0; t < 8; ++t) {
      const _Float16* wtrow = &ldsB[(colbase + t * 16 + m) * LDS_BSTRIDE];
      v8h b0 = *(const v8h*)(wtrow + kb + ka);
      v8h b1 = *(const v8h*)(wtrow + kb + 16 + ka);
      #pragma unroll
      for (int i = 0; i < 8; ++i) { bf[t][i] = b0[i]; bf[t][8 + i] = b1[i]; }
    }
    #pragma unroll
    for (int t = 0; t < 8; ++t) {
      acc[t] = __builtin_amdgcn_wmma_f32_16x16x32_f16(
          /*neg_a=*/false, afrag, /*neg_b=*/false, bf[t],
          /*c_mod=*/(short)0, acc[t], /*reuse_a=*/false, /*reuse_b=*/false);
    }
  }

  // ---- store (C layout: lanes 0-15 N=m,M=r ; lanes 16-31 N=m,M=r+8)
  #pragma unroll
  for (int t = 0; t < 8; ++t) {
    const int col = colbase + t * 16 + m;
    #pragma unroll
    for (int r = 0; r < 8; ++r) {
      const int row = row0 + r + hi * 8;
      if (row < N_NODES) Wh[(size_t)row * C_OUT + col] = acc[t][r];
    }
  }
}

// ---------------- per-node attention projections ----------------
__global__ __launch_bounds__(256) void node_score_kernel(const float* __restrict__ Wh,
                                                         const float* __restrict__ a,
                                                         float* __restrict__ s_src,
                                                         float* __restrict__ s_dst) {
  int idx = blockIdx.x * 256 + threadIdx.x;
  if (idx >= N_NODES * H_HEADS) return;
  int n = idx >> 3, h = idx & 7;
  const float* w  = Wh + (size_t)n * C_OUT + h * F_OUT;
  const float* as = a + h * A_LEN;
  const float* ad = as + F_OUT;
  float s1 = 0.f, s2 = 0.f;
  #pragma unroll
  for (int o = 0; o < F_OUT; ++o) { s1 = fmaf(w[o], as[o], s1); s2 = fmaf(w[o], ad[o], s2); }
  s_src[idx] = s1;
  s_dst[idx] = s2;
}

// ---------------- edge scores + segment max (atomicMax on uint bits, values >= 0) ----------
__global__ __launch_bounds__(256) void edge_score_kernel(const int* __restrict__ ei,
                                                         const float* __restrict__ ea,
                                                         const int* __restrict__ deg,
                                                         const float* __restrict__ s_src,
                                                         const float* __restrict__ s_dst,
                                                         const float* __restrict__ Cmat,
                                                         float* __restrict__ e_buf,
                                                         unsigned int* __restrict__ nmax) {
  int e = blockIdx.x * 256 + threadIdx.x;
  if (e >= N_EDGES) return;
  int src = ei[e], dst = ei[N_EDGES + e];
  float at[F_EDGE];
  #pragma unroll
  for (int j = 0; j < F_EDGE; ++j) at[j] = ea[(size_t)e * F_EDGE + j];
  float d = (float)deg[e];
  if (d < 1.f) d = 1.f;
  float scale = rsqrtf(d);
  #pragma unroll
  for (int h = 0; h < H_HEADS; ++h) {
    float v = s_src[src * H_HEADS + h] + s_dst[dst * H_HEADS + h];
    #pragma unroll
    for (int j = 0; j < F_EDGE; ++j) v = fmaf(at[j], Cmat[j * H_HEADS + h], v);
    v = (v > 0.f) ? v : LEAKY * v;   // leaky_relu
    v *= scale;
    e_buf[(size_t)e * H_HEADS + h] = v;
    float mv = v > 0.f ? v : 0.f;    // e_max = max(segment_max, 0) >= 0
    atomicMax(&nmax[src * H_HEADS + h], __float_as_uint(mv));
  }
}

// ---------------- exp + segment sum ----------------
__global__ __launch_bounds__(256) void edge_exp_kernel(const int* __restrict__ ei,
                                                       const float* __restrict__ nmax,
                                                       float* __restrict__ e_buf,
                                                       float* __restrict__ nsum) {
  int idx = blockIdx.x * 256 + threadIdx.x;
  if (idx >= N_EDGES * H_HEADS) return;
  int e = idx >> 3, h = idx & 7;
  int src = ei[e];
  float v = expf(e_buf[idx] - nmax[src * H_HEADS + h]);
  e_buf[idx] = v;
  atomicAdd(&nsum[src * H_HEADS + h], v);
}

// ---------------- alpha + weighted message scatter (atomic segment_sum) ----------------
__global__ __launch_bounds__(256) void edge_msg_kernel(const int* __restrict__ ei,
                                                       const float* __restrict__ e_buf,
                                                       const float* __restrict__ nsum,
                                                       const float* __restrict__ Wh,
                                                       float* __restrict__ out_acc,
                                                       float* __restrict__ alpha_out) {
  int idx = blockIdx.x * 256 + threadIdx.x;
  if (idx >= N_EDGES * H_HEADS) return;
  int e = idx >> 3, h = idx & 7;
  int src = ei[e], dst = ei[N_EDGES + e];
  float alpha = e_buf[idx] / (nsum[src * H_HEADS + h] + 1e-16f);
  alpha_out[idx] = alpha;
  const float* wrow = Wh + (size_t)dst * C_OUT + h * F_OUT;
  float* orow = out_acc + (size_t)src * C_OUT + h * F_OUT;
  #pragma unroll
  for (int o = 0; o < F_OUT; ++o) atomicAdd(&orow[o], wrow[o] * alpha);
}

// ---------------- final ELU (in place on d_out's first N*256 floats) ----------------
__global__ __launch_bounds__(256) void elu_kernel(float* __restrict__ out) {
  int i = blockIdx.x * 256 + threadIdx.x;
  if (i >= N_NODES * C_OUT) return;
  float v = out[i];
  out[i] = (v > 0.f) ? v : (expf(v) - 1.f);
}

// ---------------- launch ----------------
extern "C" void kernel_launch(void* const* d_in, const int* in_sizes, int n_in,
                              void* d_out, int out_size, void* d_ws, size_t ws_size,
                              hipStream_t stream) {
  const float* x   = (const float*)d_in[0];
  const int*   ei  = (const int*)  d_in[1];   // [2,E] src row then dst row
  const float* ea  = (const float*)d_in[2];
  const int*   deg = (const int*)  d_in[3];
  const float* W   = (const float*)d_in[4];
  const float* a   = (const float*)d_in[5];
  const float* Wew = (const float*)d_in[6];

  char* ws = (char*)d_ws;
  float*        Wh    = (float*)       (ws + OFF_WH);
  _Float16*     WT    = (_Float16*)    (ws + OFF_WT);
  float*        s_src = (float*)       (ws + OFF_SSRC);
  float*        s_dst = (float*)       (ws + OFF_SDST);
  float*        e_buf = (float*)       (ws + OFF_EBUF);
  float*        nmaxf = (float*)       (ws + OFF_NMAX);
  unsigned int* nmaxu = (unsigned int*)(ws + OFF_NMAX);
  float*        nsum  = (float*)       (ws + OFF_NSUM);
  float*        Cmat  = (float*)       (ws + OFF_CMAT);

  float* out       = (float*)d_out;                            // [N, 256]
  float* alpha_out = (float*)d_out + (size_t)N_NODES * C_OUT;  // [E, 8]

  (void)hipMemsetAsync(out,           0, (size_t)N_NODES * C_OUT * sizeof(float), stream);
  (void)hipMemsetAsync(ws + OFF_NMAX, 0, (size_t)N_NODES * H_HEADS * sizeof(float), stream);
  (void)hipMemsetAsync(ws + OFF_NSUM, 0, (size_t)N_NODES * H_HEADS * sizeof(float), stream);

  prep_kernel<<<256, 256, 0, stream>>>(W, a, Wew, WT, Cmat);

  gemm_wh_kernel<<<(N_NODES + 63) / 64, 256, 0, stream>>>(x, WT, Wh);

  node_score_kernel<<<(N_NODES * H_HEADS + 255) / 256, 256, 0, stream>>>(Wh, a, s_src, s_dst);

  edge_score_kernel<<<(N_EDGES + 255) / 256, 256, 0, stream>>>(ei, ea, deg, s_src, s_dst,
                                                               Cmat, e_buf, nmaxu);

  edge_exp_kernel<<<(N_EDGES * H_HEADS + 255) / 256, 256, 0, stream>>>(ei, nmaxf, e_buf, nsum);

  edge_msg_kernel<<<(N_EDGES * H_HEADS + 255) / 256, 256, 0, stream>>>(ei, e_buf, nsum, Wh,
                                                                       out, alpha_out);

  elu_kernel<<<(N_NODES * C_OUT + 255) / 256, 256, 0, stream>>>(out);
}
